// BlankCoder_75127567941735
// MI455X (gfx1250) — compile-verified, matrix operand order checked by reading, not run on hardware
//
#include <hip/hip_runtime.h>
#include <hip/hip_bf16.h>
#include <stdint.h>

// Problem constants (match reference)
#define BB   512
#define SS   200
#define DD   512
#define NH   8
#define DK   64
#define AA   512
#define KWIN 2
#define LW   4
#define NITER 3
#define G3D  1536   // 3*D

// Use TDM (Tensor Data Mover) for the GEMM A-tile when the builtin exists.
#if defined(__has_builtin)
#if __has_builtin(__builtin_amdgcn_tensor_load_to_lds) && __has_builtin(__builtin_amdgcn_s_wait_tensorcnt)
#define USE_TDM 1
#endif
#endif

// ---------- bf16 helpers ----------
static __device__ __forceinline__ float bf2f(unsigned short u) {
    unsigned int x = ((unsigned int)u) << 16;
    float f;
    __builtin_memcpy(&f, &x, 4);
    return f;
}
static __device__ __forceinline__ unsigned short f2bf(float f) {
    unsigned int x;
    __builtin_memcpy(&x, &f, 4);
    x += 0x7FFFu + ((x >> 16) & 1u);   // round-to-nearest-even
    return (unsigned short)(x >> 16);
}
static __device__ __forceinline__ float sigmoidf(float x) {
    return 1.0f / (1.0f + __expf(-x));
}

// ---------- WMMA types ----------
typedef __attribute__((ext_vector_type(16))) __bf16 v16bf;
typedef __attribute__((ext_vector_type(8)))  float  v8f;
union FragBF { unsigned int u[8]; v16bf v; };

#ifdef USE_TDM
typedef __attribute__((ext_vector_type(4))) unsigned int u32x4;
typedef __attribute__((ext_vector_type(8))) int i32x8;
typedef __attribute__((ext_vector_type(4))) int i32x4;
#endif

// =====================================================================
// Kernel 1: window gather, h_blk, a = tanh(h_blk@W1)@W2   (one block/batch)
// =====================================================================
__global__ __launch_bounds__(256) void k_win_attn(
    const float* __restrict__ emb, const int* __restrict__ stc_lens,
    const int* __restrict__ offs, const int* __restrict__ seps,
    const float* __restrict__ W1, const float* __restrict__ W2,
    float* __restrict__ hblk_ws, float* __restrict__ a_ws, int* __restrict__ valid_ws)
{
    __shared__ float hs[LW][DD];
    __shared__ float red[256];
    const int b = blockIdx.x;
    const int t = threadIdx.x;

    const int off = offs[b];
    const int sA = seps[b * 2 + 0];
    const int sB = seps[b * 2 + 1];
    const int len = stc_lens[b];
    const int idx = (sA < off ? 1 : 0) + (sB < off ? 1 : 0);
    const int pidx = min(max(idx - 1, 0), 1);
    const int left = (idx > 0) ? (seps[b * 2 + pidx] + 1) : 0;
    const int nidx = min(max(idx, 0), 1);
    const int right = (idx < 2) ? seps[b * 2 + nidx] : len;
    const int start = max(off - KWIN, left);
    const int end = min(off + KWIN, right);

    for (int e = t; e < LW * DD; e += 256) {
        const int l = e >> 9;
        const int d = e & (DD - 1);
        const int ind = start + l;
        const int vld = (ind < end) ? 1 : 0;
        const int ic = min(max(ind, 0), SS - 1);
        const float h = vld ? emb[((size_t)b * SS + ic) * DD + d] : 0.0f;
        hs[l][d] = h;
        hblk_ws[((size_t)b * LW + l) * DD + d] = h;
        if (d == 0) valid_ws[b * LW + l] = vld;
    }
    __syncthreads();

    float part[LW] = {0.f, 0.f, 0.f, 0.f};
    for (int ci = 0; ci < 2; ++ci) {
        const int c = t + ci * 256;
        float accl[LW] = {0.f, 0.f, 0.f, 0.f};
        for (int k = 0; k < DD; ++k) {
            const float w = W1[(size_t)k * AA + c];
#pragma unroll
            for (int l = 0; l < LW; ++l) accl[l] += hs[l][k] * w;
        }
        const float w2 = W2[c];
#pragma unroll
        for (int l = 0; l < LW; ++l) part[l] += tanhf(accl[l]) * w2;
    }
    for (int l = 0; l < LW; ++l) {
        red[t] = part[l];
        __syncthreads();
        for (int o = 128; o > 0; o >>= 1) {
            if (t < o) red[t] += red[t + o];
            __syncthreads();
        }
        if (t == 0) a_ws[b * LW + l] = red[0];
        __syncthreads();
    }
}

// =====================================================================
// Kernel 2: softmax over batch axis (axis=0) of a[B,4]  (single block)
// =====================================================================
__global__ __launch_bounds__(256) void k_softmax_batch(
    const float* __restrict__ a, float* __restrict__ s1)
{
    __shared__ float red[256];
    const int t = threadIdx.x;
    for (int l = 0; l < LW; ++l) {
        const float v0 = a[t * LW + l];
        const float v1 = a[(t + 256) * LW + l];
        red[t] = fmaxf(v0, v1);
        __syncthreads();
        for (int o = 128; o > 0; o >>= 1) {
            if (t < o) red[t] = fmaxf(red[t], red[t + o]);
            __syncthreads();
        }
        const float mx = red[0];
        __syncthreads();
        const float e0 = __expf(v0 - mx), e1 = __expf(v1 - mx);
        red[t] = e0 + e1;
        __syncthreads();
        for (int o = 128; o > 0; o >>= 1) {
            if (t < o) red[t] += red[t + o];
            __syncthreads();
        }
        const float s = red[0];
        __syncthreads();
        s1[t * LW + l] = e0 / s;
        s1[(t + 256) * LW + l] = e1 / s;
    }
}

// =====================================================================
// Kernel 3: score = softmax_l(where(valid, s1, NEG)); b0 = score . h_blk
// =====================================================================
__global__ __launch_bounds__(256) void k_score_b0(
    const float* __restrict__ s1, const int* __restrict__ valid,
    const float* __restrict__ hblk, float* __restrict__ b0)
{
    const int b = blockIdx.x;
    const int t = threadIdx.x;
    float sc[LW];
#pragma unroll
    for (int l = 0; l < LW; ++l)
        sc[l] = valid[b * LW + l] ? s1[b * LW + l] : -1.0e9f;
    const float m = fmaxf(fmaxf(sc[0], sc[1]), fmaxf(sc[2], sc[3]));
    float e[LW], sum = 0.f;
#pragma unroll
    for (int l = 0; l < LW; ++l) { e[l] = __expf(sc[l] - m); sum += e[l]; }
    const float inv = 1.0f / sum;
    for (int d = t; d < DD; d += 256) {
        float acc = 0.f;
#pragma unroll
        for (int l = 0; l < LW; ++l)
            acc += e[l] * inv * hblk[((size_t)b * LW + l) * DD + d];
        b0[(size_t)b * DD + d] = acc;
    }
}

// =====================================================================
// Generic row layer-norm over D=512, writes f32 and optional bf16 copy
// =====================================================================
__global__ __launch_bounds__(256) void k_ln_rows(
    const float* __restrict__ in, const float* __restrict__ g,
    const float* __restrict__ be, float* __restrict__ outf,
    unsigned short* __restrict__ outb)
{
    __shared__ float r1[256], r2[256];
    const int row = blockIdx.x;
    const int t = threadIdx.x;
    const float x0 = in[(size_t)row * DD + t];
    const float x1 = in[(size_t)row * DD + t + 256];
    r1[t] = x0 + x1;
    r2[t] = x0 * x0 + x1 * x1;
    __syncthreads();
    for (int o = 128; o > 0; o >>= 1) {
        if (t < o) { r1[t] += r1[t + o]; r2[t] += r2[t + o]; }
        __syncthreads();
    }
    const float mean = r1[0] * (1.0f / DD);
    const float var = r2[0] * (1.0f / DD) - mean * mean;
    const float rstd = rsqrtf(var + 1e-5f);
    const float y0 = (x0 - mean) * rstd * g[t] + be[t];
    const float y1 = (x1 - mean) * rstd * g[t + 256] + be[t + 256];
    if (outf) {
        outf[(size_t)row * DD + t] = y0;
        outf[(size_t)row * DD + t + 256] = y1;
    }
    if (outb) {
        outb[(size_t)row * DD + t] = f2bf(y0);
        outb[(size_t)row * DD + t + 256] = f2bf(y1);
    }
}

// =====================================================================
// Kernel: H = LN(embedded + pe_table[ip] + seg_emb[seg]) -> bf16
// =====================================================================
__global__ __launch_bounds__(256) void k_build_H(
    const float* __restrict__ emb, const float* __restrict__ pe,
    const float* __restrict__ seg_emb, const float* __restrict__ g,
    const float* __restrict__ be, const int* __restrict__ offs,
    const int* __restrict__ stc_lens, unsigned short* __restrict__ Hout)
{
    __shared__ float r1[256], r2[256];
    const int row = blockIdx.x;
    const int b = row / SS;
    const int s = row - b * SS;
    const int t = threadIdx.x;

    const int pos = offs[b];
    int ip = (s < pos) ? (pos - s) : (s + 1 - pos);
    if (s >= stc_lens[b]) ip = 0;
    ip = min(max(ip, 0), SS);
    const int sg = (s >= pos) ? 1 : 0;

    const size_t base = (size_t)row * DD;
    const float x0 = emb[base + t] + pe[(size_t)ip * DD + t] + seg_emb[(size_t)sg * DD + t];
    const float x1 = emb[base + t + 256] + pe[(size_t)ip * DD + t + 256] + seg_emb[(size_t)sg * DD + t + 256];
    r1[t] = x0 + x1;
    r2[t] = x0 * x0 + x1 * x1;
    __syncthreads();
    for (int o = 128; o > 0; o >>= 1) {
        if (t < o) { r1[t] += r1[t + o]; r2[t] += r2[t + o]; }
        __syncthreads();
    }
    const float mean = r1[0] * (1.0f / DD);
    const float var = r2[0] * (1.0f / DD) - mean * mean;
    const float rstd = rsqrtf(var + 1e-5f);
    Hout[base + t] = f2bf((x0 - mean) * rstd * g[t] + be[t]);
    Hout[base + t + 256] = f2bf((x1 - mean) * rstd * g[t + 256] + be[t + 256]);
}

// =====================================================================
// WMMA bf16 GEMM: C[M,N] = A[M,K] @ B[K,N] + bias
// block tile 64x64, 8 waves (4x2), wave tile 16x32 (2 acc), K-step 32.
// A tile staged to LDS by the Tensor Data Mover (wave 0 issues the D#),
// B tile staged transposed by all threads.
// =====================================================================
__global__ __launch_bounds__(256) void k_gemm_bf16(
    const unsigned short* __restrict__ A, const unsigned short* __restrict__ Bw,
    const float* __restrict__ bias, void* __restrict__ C,
    int M, int N, int K, int c_is_bf16)
{
    __shared__ unsigned short As[64][32];   // [m][k]
    __shared__ unsigned short Bs[64][32];   // [n][k]
    const int tid = threadIdx.x;
    const int lane = tid & 31;
    const int wave = tid >> 5;
    const int wm = wave & 3;        // 0..3  (M)
    const int wn = wave >> 2;       // 0..1  (N, 32 cols each)
    const int hf = (lane >> 4) & 1;
    const int lr = lane & 15;
    const int m0 = blockIdx.x * 64;
    const int n0 = blockIdx.y * 64;

    // B-tile loader coords: k = tid>>3 (0..31), 8 n's each
    const int b_k = tid >> 3;
    const int b_nc = (tid & 7) << 3;
#ifndef USE_TDM
    const int a_row = tid >> 2;
    const int a_kc = (tid & 3) << 3;
#endif

    v8f acc0 = {}, acc1 = {};
    for (int k0 = 0; k0 < K; k0 += 32) {
#ifdef USE_TDM
        // ---- A tile via TDM: 64 rows x 32 k bf16, row stride K ----
        if (wave == 0) {
            const unsigned long long ga =
                (unsigned long long)(uintptr_t)(A + (size_t)m0 * K + k0);
            const unsigned int ldsa = (unsigned int)(uintptr_t)(&As[0][0]);
            u32x4 g0;
            g0[0] = 1u;                                   // count=1, user descriptor
            g0[1] = ldsa;                                 // lds_addr
            g0[2] = (unsigned int)(ga & 0xFFFFFFFFu);     // global_addr[31:0]
            g0[3] = (unsigned int)((ga >> 32) & 0x1FFFFFFu) | 0x80000000u; // addr[56:32] | type=2
            i32x8 g1;
            g1[0] = (int)(1u << 16);                      // data_size=1 (2 bytes)
            g1[1] = (int)(((unsigned)K & 0xFFFFu) << 16); // tensor_dim0[15:0]
            g1[2] = (int)((((unsigned)K >> 16) & 0xFFFFu) |
                          (((unsigned)M & 0xFFFFu) << 16));       // tdim0 hi | tdim1 lo
            g1[3] = (int)((((unsigned)M >> 16) & 0xFFFFu) |
                          (32u << 16));                           // tdim1 hi | tile_dim0=32
            g1[4] = (int)64u;                             // tile_dim1=64, tile_dim2=0
            g1[5] = (int)(unsigned)K;                     // tensor_dim0_stride[31:0]
            g1[6] = 0;                                    // stride0 hi | stride1 lo
            g1[7] = 0;                                    // stride1 hi
            i32x4 g2 = {0, 0, 0, 0};
            i32x4 g3 = {0, 0, 0, 0};
            i32x8 g4 = {0, 0, 0, 0, 0, 0, 0, 0};
            __builtin_amdgcn_tensor_load_to_lds(g0, g1, g2, g3, g4, 0);
        }
#else
        const uint4 av = *reinterpret_cast<const uint4*>(
            A + (size_t)(m0 + a_row) * K + k0 + a_kc);
        *reinterpret_cast<uint4*>(&As[a_row][a_kc]) = av;
#endif
        // ---- B tile: 8 bf16 per thread, transposed store to [n][k] ----
        const uint4 bv4 = *reinterpret_cast<const uint4*>(
            Bw + (size_t)(k0 + b_k) * N + n0 + b_nc);
        const unsigned short* bsrc = reinterpret_cast<const unsigned short*>(&bv4);
#pragma unroll
        for (int j = 0; j < 8; ++j) Bs[b_nc + j][b_k] = bsrc[j];
        // prefetch next B tile
        __builtin_prefetch(Bw + (size_t)(k0 + 32 + b_k) * N + n0 + b_nc, 0, 1);
#ifdef USE_TDM
        if (wave == 0) __builtin_amdgcn_s_wait_tensorcnt((short)0);
#endif
        __syncthreads();

        // ---- fragments per ISA 16-bit A/B lane layout ----
        FragBF fa, fb0, fb1;
#pragma unroll
        for (int v = 0; v < 8; ++v) {
            const int kk = (v < 4 ? (v << 1) : 16 + ((v - 4) << 1)) + (hf << 3);
            fa.u[v] = *reinterpret_cast<const unsigned int*>(&As[wm * 16 + lr][kk]);
            fb0.u[v] = *reinterpret_cast<const unsigned int*>(&Bs[wn * 32 + lr][kk]);
            fb1.u[v] = *reinterpret_cast<const unsigned int*>(&Bs[wn * 32 + 16 + lr][kk]);
        }
        acc0 = __builtin_amdgcn_wmma_f32_16x16x32_bf16(
            false, fa.v, false, fb0.v, (short)0, acc0, false, false);
        acc1 = __builtin_amdgcn_wmma_f32_16x16x32_bf16(
            false, fa.v, false, fb1.v, (short)0, acc1, false, false);
        __syncthreads();
    }

    // epilogue: C layout  M = v + 8*hf, N = lr
    const int col0 = n0 + wn * 32 + lr;
    const int col1 = col0 + 16;
    const float bv0 = bias ? bias[col0] : 0.0f;
    const float bv1 = bias ? bias[col1] : 0.0f;
#pragma unroll
    for (int v = 0; v < 8; ++v) {
        const int row = m0 + wm * 16 + v + (hf << 3);
        const float val0 = acc0[v] + bv0;
        const float val1 = acc1[v] + bv1;
        if (c_is_bf16) {
            ((unsigned short*)C)[(size_t)row * N + col0] = f2bf(val0);
            ((unsigned short*)C)[(size_t)row * N + col1] = f2bf(val1);
        } else {
            ((float*)C)[(size_t)row * N + col0] = val0;
            ((float*)C)[(size_t)row * N + col1] = val1;
        }
    }
}

// =====================================================================
// Attention: one wave per (b,h); sigmoid scores over S, accumulate V
// =====================================================================
__global__ __launch_bounds__(32) void k_attn(
    const float* __restrict__ q, const unsigned short* __restrict__ kbf,
    const unsigned short* __restrict__ vbf, const int* __restrict__ stc_lens,
    float* __restrict__ m_raw)
{
    const int b = blockIdx.x / NH;
    const int h = blockIdx.x - b * NH;
    const int lane = threadIdx.x;
    const int d0 = lane, d1 = lane + 32;
    const float scale = 0.125f;  // 1/sqrt(64)

    const float q0 = q[(size_t)b * DD + h * DK + d0];
    const float q1 = q[(size_t)b * DD + h * DK + d1];
    const int len = stc_lens[b];
    float acc0 = 0.f, acc1 = 0.f;

    for (int s = 0; s < SS; ++s) {
        const size_t base = ((size_t)b * SS + s) * DD + h * DK;
        float p = q0 * bf2f(kbf[base + d0]) + q1 * bf2f(kbf[base + d1]);
#pragma unroll
        for (int o = 16; o > 0; o >>= 1) p += __shfl_xor(p, o, 32);
        const float pa = (s < len) ? sigmoidf(p * scale) : 0.0f;
        acc0 += pa * bf2f(vbf[base + d0]);
        acc1 += pa * bf2f(vbf[base + d1]);
    }
    m_raw[(size_t)b * DD + h * DK + d0] = acc0;
    m_raw[(size_t)b * DD + h * DK + d1] = acc1;
}

// =====================================================================
// GRU elementwise update, writes b_t (f32) and bf16 copy
// =====================================================================
__global__ __launch_bounds__(256) void k_gru(
    const float* __restrict__ gi, const float* __restrict__ gh,
    float* __restrict__ bt, unsigned short* __restrict__ btb)
{
    const int i = blockIdx.x * 256 + threadIdx.x;
    const int b = i >> 9;
    const int d = i & (DD - 1);
    const size_t gb = (size_t)b * G3D;
    const float ir = gi[gb + d], iz = gi[gb + DD + d], inn = gi[gb + 2 * DD + d];
    const float hr = gh[gb + d], hz = gh[gb + DD + d], hn = gh[gb + 2 * DD + d];
    const float r = sigmoidf(ir + hr);
    const float z = sigmoidf(iz + hz);
    const float n = tanhf(inn + r * hn);
    const float old = bt[i];
    const float nb = (1.0f - z) * n + z * old;
    bt[i] = nb;
    btb[i] = f2bf(nb);
}

// ---------- weight casts ----------
__global__ void k_cast_bf(const float* __restrict__ src, unsigned short* __restrict__ dst, int n)
{
    const int i = blockIdx.x * 256 + threadIdx.x;
    if (i < n) dst[i] = f2bf(src[i]);
}
// src [rows][cols] -> dst [cols][rows]  (dst[k*rows + j] = src[j*cols + k])
__global__ void k_castT_bf(const float* __restrict__ src, unsigned short* __restrict__ dst,
                           int rows, int cols)
{
    const int i = blockIdx.x * 256 + threadIdx.x;
    if (i < rows * cols) {
        const int j = i / cols;
        const int k = i - j * cols;
        dst[(size_t)k * rows + j] = f2bf(src[i]);
    }
}

// =====================================================================
// Host launch
// =====================================================================
extern "C" void kernel_launch(void* const* d_in, const int* in_sizes, int n_in,
                              void* d_out, int out_size, void* d_ws, size_t ws_size,
                              hipStream_t stream)
{
    const float* embedded = (const float*)d_in[0];
    const int* stc_lens = (const int*)d_in[1];
    const int* offsets = (const int*)d_in[2];
    const int* sep_lst = (const int*)d_in[3];
    const float* W1 = (const float*)d_in[4];
    const float* W2 = (const float*)d_in[5];
    const float* ln_g = (const float*)d_in[6];
    const float* ln_b = (const float*)d_in[7];
    const float* lng_g = (const float*)d_in[8];
    const float* lng_b = (const float*)d_in[9];
    const float* Wq = (const float*)d_in[10];
    const float* bq = (const float*)d_in[11];
    const float* Wk = (const float*)d_in[12];
    const float* bk = (const float*)d_in[13];
    const float* Wv = (const float*)d_in[14];
    const float* bv = (const float*)d_in[15];
    const float* W_ih = (const float*)d_in[16];
    const float* W_hh = (const float*)d_in[17];
    const float* b_ih = (const float*)d_in[18];
    const float* b_hh = (const float*)d_in[19];
    const float* seg_emb = (const float*)d_in[20];
    const float* pe_table = (const float*)d_in[21];

    // workspace carving
    char* p = (char*)d_ws;
    auto alloc = [&](size_t bytes) -> void* {
        void* r = (void*)p;
        p += (bytes + 255) & ~(size_t)255;
        return r;
    };
    float* ws_hblk = (float*)alloc((size_t)BB * LW * DD * 4);
    float* ws_a = (float*)alloc((size_t)BB * LW * 4);
    int* ws_valid = (int*)alloc((size_t)BB * LW * 4);
    float* ws_s1 = (float*)alloc((size_t)BB * LW * 4);
    float* ws_b0 = (float*)alloc((size_t)BB * DD * 4);
    float* ws_bt = (float*)alloc((size_t)BB * DD * 4);
    unsigned short* ws_bt_bf = (unsigned short*)alloc((size_t)BB * DD * 2);
    unsigned short* ws_H = (unsigned short*)alloc((size_t)BB * SS * DD * 2);
    unsigned short* ws_Kp = (unsigned short*)alloc((size_t)BB * SS * DD * 2);
    unsigned short* ws_Vp = (unsigned short*)alloc((size_t)BB * SS * DD * 2);
    unsigned short* ws_Wq = (unsigned short*)alloc((size_t)DD * DD * 2);
    unsigned short* ws_Wk = (unsigned short*)alloc((size_t)DD * DD * 2);
    unsigned short* ws_Wv = (unsigned short*)alloc((size_t)DD * DD * 2);
    unsigned short* ws_WihT = (unsigned short*)alloc((size_t)DD * G3D * 2);
    unsigned short* ws_WhhT = (unsigned short*)alloc((size_t)DD * G3D * 2);
    float* ws_q = (float*)alloc((size_t)BB * DD * 4);
    float* ws_mraw = (float*)alloc((size_t)BB * DD * 4);
    float* ws_mln = (float*)alloc((size_t)BB * DD * 4);
    unsigned short* ws_m_bf = (unsigned short*)alloc((size_t)BB * DD * 2);
    float* ws_gi = (float*)alloc((size_t)BB * G3D * 4);
    float* ws_gh = (float*)alloc((size_t)BB * G3D * 4);

    // 1-3: window attention -> b0
    k_win_attn<<<BB, 256, 0, stream>>>(embedded, stc_lens, offsets, sep_lst,
                                       W1, W2, ws_hblk, ws_a, ws_valid);
    k_softmax_batch<<<1, 256, 0, stream>>>(ws_a, ws_s1);
    k_score_b0<<<BB, 256, 0, stream>>>(ws_s1, ws_valid, ws_hblk, ws_b0);
    // b_t = LN(b0) (pe_table[0] row is zeros)
    k_ln_rows<<<BB, 256, 0, stream>>>(ws_b0, ln_g, ln_b, ws_bt, ws_bt_bf);

    // weight casts
    k_cast_bf<<<(DD * DD + 255) / 256, 256, 0, stream>>>(Wq, ws_Wq, DD * DD);
    k_cast_bf<<<(DD * DD + 255) / 256, 256, 0, stream>>>(Wk, ws_Wk, DD * DD);
    k_cast_bf<<<(DD * DD + 255) / 256, 256, 0, stream>>>(Wv, ws_Wv, DD * DD);
    k_castT_bf<<<(G3D * DD + 255) / 256, 256, 0, stream>>>(W_ih, ws_WihT, G3D, DD);
    k_castT_bf<<<(G3D * DD + 255) / 256, 256, 0, stream>>>(W_hh, ws_WhhT, G3D, DD);

    // H
    k_build_H<<<BB * SS, 256, 0, stream>>>(embedded, pe_table, seg_emb,
                                           ln_g, ln_b, offsets, stc_lens, ws_H);

    // K/V projections (M=102400, N=K=512), bf16 out
    dim3 gkv(BB * SS / 64, DD / 64);
    k_gemm_bf16<<<gkv, 256, 0, stream>>>(ws_H, ws_Wk, bk, ws_Kp, BB * SS, DD, DD, 1);
    k_gemm_bf16<<<gkv, 256, 0, stream>>>(ws_H, ws_Wv, bv, ws_Vp, BB * SS, DD, DD, 1);

    // recurrence
    dim3 gq(BB / 64, DD / 64);
    dim3 gg(BB / 64, G3D / 64);
    for (int it = 0; it < NITER; ++it) {
        k_gemm_bf16<<<gq, 256, 0, stream>>>(ws_bt_bf, ws_Wq, bq, ws_q, BB, DD, DD, 0);
        k_attn<<<BB * NH, 32, 0, stream>>>(ws_q, ws_Kp, ws_Vp, stc_lens, ws_mraw);
        k_ln_rows<<<BB, 256, 0, stream>>>(ws_mraw, lng_g, lng_b, ws_mln, ws_m_bf);
        k_gemm_bf16<<<gg, 256, 0, stream>>>(ws_m_bf, ws_WihT, b_ih, ws_gi, BB, G3D, DD, 0);
        k_gemm_bf16<<<gg, 256, 0, stream>>>(ws_bt_bf, ws_WhhT, b_hh, ws_gh, BB, G3D, DD, 0);
        k_gru<<<(BB * DD) / 256, 256, 0, stream>>>(ws_gi, ws_gh, ws_bt, ws_bt_bf);
    }

    (void)hipMemcpyAsync(d_out, ws_bt, (size_t)BB * DD * sizeof(float),
                         hipMemcpyDeviceToDevice, stream);
    (void)in_sizes; (void)n_in; (void)out_size; (void)ws_size;
}
